// NavVanillaTransformer_51075751084194
// MI455X (gfx1250) — matrix operand order repaired and checked by reading, not compiled, not run on hardware
//
#include <hip/hip_runtime.h>

// ---------------------------------------------------------------------------
// NavVanillaTransformer on MI455X (gfx1250)
//   - all GEMMs + attention matmuls on v_wmma_f32_16x16x32_bf16
//   - weights pre-cast to bf16 and pre-transposed [N][K] once per launch
//   - GEMM tiles staged with GLOBAL_LOAD_ASYNC_TO_LDS_B128, double-buffered LDS
//   - fp32 softmax / LayerNorm / GELU epilogues
// ---------------------------------------------------------------------------
#define BB   2
#define TT   2048
#define HH   1024
#define NHH  16
#define DD   64
#define FFF  4096
#define LL   6
#define BT   (BB * TT)   // 4096 rows

typedef __attribute__((ext_vector_type(16))) __bf16 v16bf;
typedef __attribute__((ext_vector_type(8)))  float  v8f;

union FragBF {
    v16bf    v;
    unsigned u[8];
};

__device__ __forceinline__ unsigned short f2bf1(float f) {
    union { float f; unsigned u; } a;
    a.f = f;
    unsigned u = a.u;
    unsigned r = u + 0x7FFFu + ((u >> 16) & 1u);   // round-to-nearest-even
    return (unsigned short)(r >> 16);
}

// A-fragment (16-bit 16x32) interleaved K pattern: VGPR i holds K = base, base+1
__device__ __forceinline__ int a_kbase(int i, int half) {
    return 2 * i + ((i >= 4) ? 8 : 0) + 8 * half;
}
// B-fragment (16-bit 32x16) interleaved K pattern (same as A)
__device__ __forceinline__ int b_kbase(int i, int half) {
    return 2 * i + ((i >= 4) ? 8 : 0) + 8 * half;
}

__device__ __forceinline__ v8f wmma_bf16(const FragBF& a, const FragBF& b, v8f c) {
    return __builtin_amdgcn_wmma_f32_16x16x32_bf16(
        false, a.v, false, b.v, (short)0, c, false, false);
}

// ---------------- CDNA5 async global->LDS copy (guarded) -------------------
#if defined(__HIP_DEVICE_COMPILE__) && \
    __has_builtin(__builtin_amdgcn_global_load_async_to_lds_b128)
#define HAVE_ASYNC 1
#else
#define HAVE_ASYNC 0
#endif

#if HAVE_ASYNC
#define AS1 __attribute__((address_space(1)))
#define AS3 __attribute__((address_space(3)))
typedef int v4i __attribute__((vector_size(16)));
__device__ __forceinline__ void async_cp16(const void* g, void* l) {
    __builtin_amdgcn_global_load_async_to_lds_b128(
        (AS1 v4i*)(unsigned long long)(uintptr_t)g,
        (AS3 v4i*)(unsigned)(uintptr_t)l, 0, 0);
}
#endif

__device__ __forceinline__ void wait_async0() {
#if defined(__HIP_DEVICE_COMPILE__) && __has_builtin(__builtin_amdgcn_s_wait_asynccnt)
    __builtin_amdgcn_s_wait_asynccnt(0);
#else
    asm volatile("s_wait_asynccnt 0" ::: "memory");
#endif
}

// ---------------------------------------------------------------------------
// one-time casts: fp32 -> bf16, and fp32 [K][N] -> bf16 transposed [N][K]
// ---------------------------------------------------------------------------
__global__ void __launch_bounds__(256)
cast_bf16(const float* __restrict__ s, unsigned short* __restrict__ d, int n)
{
    int i = blockIdx.x * 256 + threadIdx.x;
    if (i < n) d[i] = f2bf1(s[i]);
}

__global__ void __launch_bounds__(256)
cast_transpose(const float* __restrict__ s, unsigned short* __restrict__ d,
               int K, int N, int total)
{
    int i = blockIdx.x * 256 + threadIdx.x;
    if (i < total) {
        int per = K * N;
        int l = i / per, r = i % per;
        int n = r / K, k = r % K;
        d[(size_t)l * per + r] = f2bf1(s[(size_t)l * per + (size_t)k * N + n]);
    }
}

// ---------------------------------------------------------------------------
// GEMM: out[M,N] = A[M,K] @ B[K,N] + bias, A bf16 [M][K], Bt bf16 [N][K]
// 128x128x32 block tile, 256 threads = 8 waves (2x4), wave = 64x32 (4x2 tiles)
// Tiles staged with async b128 copies, double-buffered LDS.
// ---------------------------------------------------------------------------
template <bool GELU, bool BF16OUT>
__global__ void __launch_bounds__(256)
gemm_bf16_wmma(const unsigned short* __restrict__ A,
               const unsigned short* __restrict__ Bt,
               const float* __restrict__ bias, void* __restrict__ out,
               int M, int N, int K)
{
    constexpr int BM = 128, BN = 128, BK = 32;
#if HAVE_ASYNC
    __shared__ unsigned short sA[2][BM * BK];
    __shared__ unsigned short sB[2][BN * BK];
#else
    __shared__ unsigned short sA[1][BM * BK];
    __shared__ unsigned short sB[1][BN * BK];
#endif

    const int tid  = threadIdx.x;
    const int lane = tid & 31;
    const int wave = tid >> 5;
    const int wm   = wave >> 2;
    const int wn   = wave & 3;
    const int half = lane >> 4;
    const int l16  = lane & 15;
    const int m0   = blockIdx.y * BM;
    const int n0   = blockIdx.x * BN;

    v8f acc[4][2];
#pragma unroll
    for (int mi = 0; mi < 4; ++mi)
#pragma unroll
        for (int ni = 0; ni < 2; ++ni) acc[mi][ni] = {};

    const int nK = K / BK;

#if HAVE_ASYNC
    // prologue: stage tile 0 into buffer 0 (4 async b128 ops per thread/stage)
    {
#pragma unroll
        for (int i = 0; i < 2; ++i) {
            int c = tid + i * 256;               // 512 16B chunks per 8KB tile
            int row = c >> 2, col = (c & 3) * 8; // 4 chunks per 64B row
            async_cp16(A + (size_t)(m0 + row) * K + col, &sA[0][c * 8]);
            async_cp16(Bt + (size_t)(n0 + row) * K + col, &sB[0][c * 8]);
        }
    }
#endif

    for (int kt = 0; kt < nK; ++kt) {
#if HAVE_ASYNC
        const int cb = kt & 1;
        wait_async0();           // this wave's stage-kt copies landed in LDS
        __syncthreads();         // everyone's copies landed; prev reads done
        if (kt + 1 < nK) {
            const int k0n = (kt + 1) * BK;
#pragma unroll
            for (int i = 0; i < 2; ++i) {
                int c = tid + i * 256;
                int row = c >> 2, col = (c & 3) * 8;
                async_cp16(A + (size_t)(m0 + row) * K + k0n + col,
                           &sA[cb ^ 1][c * 8]);
                async_cp16(Bt + (size_t)(n0 + row) * K + k0n + col,
                           &sB[cb ^ 1][c * 8]);
            }
        }
#else
        const int cb = 0;
        const int k0 = kt * BK;
        __syncthreads();
#pragma unroll
        for (int i = 0; i < 2; ++i) {
            int c = tid + i * 256;
            int row = c >> 2, col = (c & 3) * 8;
            *(uint4*)&sA[0][c * 8] =
                *(const uint4*)(A + (size_t)(m0 + row) * K + k0 + col);
            *(uint4*)&sB[0][c * 8] =
                *(const uint4*)(Bt + (size_t)(n0 + row) * K + k0 + col);
        }
        if (kt + 1 < nK)
            __builtin_prefetch(&Bt[(size_t)(n0 + (tid >> 1)) * K + k0 + BK], 0, 3);
        __syncthreads();
#endif

        FragBF af[4];
#pragma unroll
        for (int mi = 0; mi < 4; ++mi) {
            int row = wm * 64 + mi * 16 + l16;
#pragma unroll
            for (int i = 0; i < 8; ++i)
                af[mi].u[i] =
                    *(const unsigned*)&sA[cb][row * BK + a_kbase(i, half)];
        }
        FragBF bf[2];
#pragma unroll
        for (int ni = 0; ni < 2; ++ni) {
            int col = wn * 32 + ni * 16 + l16;
#pragma unroll
            for (int i = 0; i < 8; ++i)
                bf[ni].u[i] =
                    *(const unsigned*)&sB[cb][col * BK + b_kbase(i, half)];
        }
#pragma unroll
        for (int mi = 0; mi < 4; ++mi)
#pragma unroll
            for (int ni = 0; ni < 2; ++ni)
                acc[mi][ni] = wmma_bf16(af[mi], bf[ni], acc[mi][ni]);
    }

#pragma unroll
    for (int mi = 0; mi < 4; ++mi)
#pragma unroll
        for (int ni = 0; ni < 2; ++ni)
#pragma unroll
            for (int v = 0; v < 8; ++v) {
                int r = m0 + wm * 64 + mi * 16 + (half << 3) + v;
                int c = n0 + wn * 32 + ni * 16 + l16;
                float x = acc[mi][ni][v] + bias[c];
                if (GELU) x = 0.5f * x * (1.0f + erff(x * 0.70710678118654752f));
                if (BF16OUT)
                    ((unsigned short*)out)[(size_t)r * N + c] = f2bf1(x);
                else
                    ((float*)out)[(size_t)r * N + c] = x;
            }
}

// ---------------------------------------------------------------------------
// Flash attention: 4 waves/block, wave owns 16 query rows, streams 32-key tiles.
// q,k,v bf16 [B,T,H], head h at cols h*D..h*D+63. Output bf16 (feeds proj GEMM).
// ---------------------------------------------------------------------------
__global__ void __launch_bounds__(128)
attn_flash(const unsigned short* __restrict__ Q,
           const unsigned short* __restrict__ Kb,
           const unsigned short* __restrict__ V,
           unsigned short* __restrict__ O)
{
    __shared__ unsigned short pl[4][16 * 32];

    const int lane = threadIdx.x & 31;
    const int wave = threadIdx.x >> 5;
    const int half = lane >> 4;
    const int l16  = lane & 15;

    const int tilesPerSeq = TT / 64;
    const int bh = blockIdx.x / tilesPerSeq;
    const int qt = (blockIdx.x % tilesPerSeq) * 64 + wave * 16;
    const int b  = bh / NHH;
    const int h  = bh % NHH;

    const unsigned short* qp = Q  + (size_t)b * TT * HH + h * DD;
    const unsigned short* kp = Kb + (size_t)b * TT * HH + h * DD;
    const unsigned short* vp = V  + (size_t)b * TT * HH + h * DD;

    FragBF qf[2];
    const int qrow = qt + l16;
#pragma unroll
    for (int dA = 0; dA < 2; ++dA)
#pragma unroll
        for (int i = 0; i < 8; ++i)
            qf[dA].u[i] = *(const unsigned*)&qp[(size_t)qrow * HH + dA * 32 +
                                               a_kbase(i, half)];

    v8f o[4];
#pragma unroll
    for (int nt = 0; nt < 4; ++nt) o[nt] = {};
    float mrow[8], lrow[8];
#pragma unroll
    for (int v = 0; v < 8; ++v) { mrow[v] = -1e30f; lrow[v] = 0.0f; }

    const int nkt = (qt + 16 + 31) >> 5;
    for (int t = 0; t < nkt; ++t) {
        const int kt0 = t * 32;

        FragBF kf[2][2];
#pragma unroll
        for (int nt = 0; nt < 2; ++nt) {
            int key = kt0 + nt * 16 + l16;
#pragma unroll
            for (int dA = 0; dA < 2; ++dA)
#pragma unroll
                for (int i = 0; i < 8; ++i)
                    kf[nt][dA].u[i] =
                        *(const unsigned*)&kp[(size_t)key * HH + dA * 32 +
                                              b_kbase(i, half)];
        }
        v8f s0 = {}, s1 = {};
        s0 = wmma_bf16(qf[0], kf[0][0], s0);
        s0 = wmma_bf16(qf[1], kf[0][1], s0);
        s1 = wmma_bf16(qf[0], kf[1][0], s1);
        s1 = wmma_bf16(qf[1], kf[1][1], s1);

        float p0[8], p1[8];
#pragma unroll
        for (int v = 0; v < 8; ++v) {
            int tq = qt + (half << 3) + v;
            float a = s0[v] * 0.125f;
            float c = s1[v] * 0.125f;
            if (kt0 + l16      > tq) a = -1e30f;
            if (kt0 + 16 + l16 > tq) c = -1e30f;
            float rm = fmaxf(a, c);
#pragma unroll
            for (int off = 8; off >= 1; off >>= 1)
                rm = fmaxf(rm, __shfl_xor(rm, off, 32));
            float mn = fmaxf(mrow[v], rm);
            float al = __expf(mrow[v] - mn);
            float e0 = __expf(a - mn);
            float e1 = __expf(c - mn);
            float rs = e0 + e1;
#pragma unroll
            for (int off = 8; off >= 1; off >>= 1)
                rs += __shfl_xor(rs, off, 32);
            lrow[v] = lrow[v] * al + rs;
            mrow[v] = mn;
            p0[v] = e0; p1[v] = e1;
#pragma unroll
            for (int nt = 0; nt < 4; ++nt) o[nt][v] *= al;
        }

        unsigned short* pw = pl[wave];
#pragma unroll
        for (int v = 0; v < 8; ++v) {
            int m = (half << 3) + v;
            pw[m * 32 + l16]      = f2bf1(p0[v]);
            pw[m * 32 + 16 + l16] = f2bf1(p1[v]);
        }
        asm volatile("s_wait_dscnt 0" ::: "memory");
        __builtin_amdgcn_wave_barrier();

        FragBF pf;
#pragma unroll
        for (int i = 0; i < 8; ++i)
            pf.u[i] = *(const unsigned*)&pw[l16 * 32 + a_kbase(i, half)];

#pragma unroll
        for (int nt = 0; nt < 4; ++nt) {
            FragBF vf;
            int d = nt * 16 + l16;
#pragma unroll
            for (int i = 0; i < 8; ++i) {
                int kk = b_kbase(i, half);
                unsigned lo = vp[(size_t)(kt0 + kk)     * HH + d];
                unsigned hi = vp[(size_t)(kt0 + kk + 1) * HH + d];
                vf.u[i] = lo | (hi << 16);
            }
            o[nt] = wmma_bf16(pf, vf, o[nt]);
        }
    }

#pragma unroll
    for (int nt = 0; nt < 4; ++nt)
#pragma unroll
        for (int v = 0; v < 8; ++v) {
            int r = qt + (half << 3) + v;
            int c = h * DD + nt * 16 + l16;
            O[((size_t)b * TT + r) * HH + c] = f2bf1(o[nt][v] / lrow[v]);
        }
}

// ---------------------------------------------------------------------------
// out[r,:] = LayerNorm(X[r,:] + R[r,:]) * g + be ; also bf16 copy for GEMMs
// ---------------------------------------------------------------------------
__global__ void __launch_bounds__(256)
residual_ln(const float* __restrict__ X, const float* __restrict__ R,
            const float* __restrict__ g, const float* __restrict__ be,
            float* __restrict__ out, unsigned short* __restrict__ outb)
{
    __shared__ float red[8];
    const int r    = blockIdx.x;
    const int tid  = threadIdx.x;
    const int lane = tid & 31;
    const int wave = tid >> 5;
    const float* x = X + (size_t)r * HH;
    const float* o = R + (size_t)r * HH;

    float vals[4];
    float s = 0.0f;
#pragma unroll
    for (int i = 0; i < 4; ++i) {
        vals[i] = x[tid + i * 256] + o[tid + i * 256];
        s += vals[i];
    }
#pragma unroll
    for (int off = 16; off >= 1; off >>= 1) s += __shfl_xor(s, off, 32);
    if (lane == 0) red[wave] = s;
    __syncthreads();
    float tot = 0.0f;
#pragma unroll
    for (int w = 0; w < 8; ++w) tot += red[w];
    const float mean = tot * (1.0f / HH);

    float vs = 0.0f;
#pragma unroll
    for (int i = 0; i < 4; ++i) {
        float d = vals[i] - mean;
        vs += d * d;
    }
#pragma unroll
    for (int off = 16; off >= 1; off >>= 1) vs += __shfl_xor(vs, off, 32);
    __syncthreads();
    if (lane == 0) red[wave] = vs;
    __syncthreads();
    float totv = 0.0f;
#pragma unroll
    for (int w = 0; w < 8; ++w) totv += red[w];
    const float inv = rsqrtf(totv * (1.0f / HH) + 1e-5f);

#pragma unroll
    for (int i = 0; i < 4; ++i) {
        int c = tid + i * 256;
        float y = (vals[i] - mean) * inv * g[c] + be[c];
        out[(size_t)r * HH + c] = y;
        outb[(size_t)r * HH + c] = f2bf1(y);
    }
}

// ---------------------------------------------------------------------------
extern "C" void kernel_launch(void* const* d_in, const int* in_sizes, int n_in,
                              void* d_out, int out_size, void* d_ws, size_t ws_size,
                              hipStream_t stream)
{
    const float* x   = (const float*)d_in[0];
    const float* Wq  = (const float*)d_in[1];
    const float* bq  = (const float*)d_in[2];
    const float* Wk  = (const float*)d_in[3];
    const float* bk  = (const float*)d_in[4];
    const float* Wv  = (const float*)d_in[5];
    const float* bv  = (const float*)d_in[6];
    const float* Wp  = (const float*)d_in[7];
    const float* bp  = (const float*)d_in[8];
    const float* W1  = (const float*)d_in[9];
    const float* b1  = (const float*)d_in[10];
    const float* W2  = (const float*)d_in[11];
    const float* b2  = (const float*)d_in[12];
    const float* g1  = (const float*)d_in[13];
    const float* be1 = (const float*)d_in[14];
    const float* g2  = (const float*)d_in[15];
    const float* be2 = (const float*)d_in[16];

    char* ws = (char*)d_ws;
    size_t off = 0;
    const size_t BTH = (size_t)BT * HH;          // 4M activations
    const size_t WHH = (size_t)LL * HH * HH;     // 6M weights per H x H type
    const size_t WHF = (size_t)LL * HH * FFF;    // 25M weights per H x FF type

    float*          cur   = (float*)(ws + off);          off += BTH * 4;
    float*          tmp   = (float*)(ws + off);          off += BTH * 4;
    unsigned short* curb  = (unsigned short*)(ws + off); off += BTH * 2;
    unsigned short* xb    = (unsigned short*)(ws + off); off += BTH * 2;
    unsigned short* qb    = (unsigned short*)(ws + off); off += BTH * 2;
    unsigned short* kb    = (unsigned short*)(ws + off); off += BTH * 2;
    unsigned short* vb    = (unsigned short*)(ws + off); off += BTH * 2;
    unsigned short* attnb = (unsigned short*)(ws + off); off += BTH * 2;
    unsigned short* ffhb  = (unsigned short*)(ws + off); off += (size_t)BT * FFF * 2;
    unsigned short* WqT   = (unsigned short*)(ws + off); off += WHH * 2;
    unsigned short* WkT   = (unsigned short*)(ws + off); off += WHH * 2;
    unsigned short* WvT   = (unsigned short*)(ws + off); off += WHH * 2;
    unsigned short* WpT   = (unsigned short*)(ws + off); off += WHH * 2;
    unsigned short* W1T   = (unsigned short*)(ws + off); off += WHF * 2;
    unsigned short* W2T   = (unsigned short*)(ws + off); off += WHF * 2;

    const dim3 blk(256);
    // one-time precision/layout prep
    cast_bf16<<<dim3((int)(BTH / 256)), blk, 0, stream>>>(x, xb, (int)BTH);
    cast_transpose<<<dim3((int)(WHH / 256)), blk, 0, stream>>>(Wq, WqT, HH, HH, (int)WHH);
    cast_transpose<<<dim3((int)(WHH / 256)), blk, 0, stream>>>(Wk, WkT, HH, HH, (int)WHH);
    cast_transpose<<<dim3((int)(WHH / 256)), blk, 0, stream>>>(Wv, WvT, HH, HH, (int)WHH);
    cast_transpose<<<dim3((int)(WHH / 256)), blk, 0, stream>>>(Wp, WpT, HH, HH, (int)WHH);
    cast_transpose<<<dim3((int)(WHF / 256)), blk, 0, stream>>>(W1, W1T, HH, FFF, (int)WHF);
    cast_transpose<<<dim3((int)(WHF / 256)), blk, 0, stream>>>(W2, W2T, FFF, HH, (int)WHF);

    const dim3 gQKV(HH / 128, BT / 128);
    const dim3 gFC1(FFF / 128, BT / 128);
    const dim3 gAttn(BB * NHH * (TT / 64));

    const float* xcur = x;           // fp32 residual stream
    const unsigned short* xcb = xb;  // bf16 copy for GEMM A-side
    for (int l = 0; l < LL; ++l) {
        const size_t wHH = (size_t)l * HH * HH;
        const size_t wHF = (size_t)l * HH * FFF;

        gemm_bf16_wmma<false, true><<<gQKV, blk, 0, stream>>>(
            xcb, WqT + wHH, bq + (size_t)l * HH, qb, BT, HH, HH);
        gemm_bf16_wmma<false, true><<<gQKV, blk, 0, stream>>>(
            xcb, WkT + wHH, bk + (size_t)l * HH, kb, BT, HH, HH);
        gemm_bf16_wmma<false, true><<<gQKV, blk, 0, stream>>>(
            xcb, WvT + wHH, bv + (size_t)l * HH, vb, BT, HH, HH);

        attn_flash<<<gAttn, dim3(128), 0, stream>>>(qb, kb, vb, attnb);

        gemm_bf16_wmma<false, false><<<gQKV, blk, 0, stream>>>(
            attnb, WpT + wHH, bp + (size_t)l * HH, tmp, BT, HH, HH);

        residual_ln<<<dim3(BT), blk, 0, stream>>>(
            xcur, tmp, g1 + (size_t)l * HH, be1 + (size_t)l * HH, cur, curb);

        gemm_bf16_wmma<true, true><<<gFC1, blk, 0, stream>>>(
            curb, W1T + wHF, b1 + (size_t)l * FFF, ffhb, BT, FFF, HH);
        gemm_bf16_wmma<false, false><<<gQKV, blk, 0, stream>>>(
            ffhb, W2T + wHF, b2 + (size_t)l * HH, tmp, BT, HH, FFF);

        float* lnOut = (l == LL - 1) ? (float*)d_out : cur;
        residual_ln<<<dim3(BT), blk, 0, stream>>>(
            cur, tmp, g2 + (size_t)l * HH, be2 + (size_t)l * HH, lnOut, curb);

        xcur = cur;
        xcb  = curb;
    }
    (void)in_sizes; (void)n_in; (void)out_size; (void)ws_size;
}
